// MultiHeadAttention_14912126451818
// MI455X (gfx1250) — matrix operand order repaired and checked by reading, not compiled, not run on hardware
//
#include <hip/hip_runtime.h>
#include <hip/hip_bf16.h>

#define B_   2
#define S_   2048
#define D_   2048
#define H_   16
#define KVH_ 8
#define HD_  128

typedef unsigned short u16;
typedef __attribute__((ext_vector_type(4)))  unsigned int u32x4;
typedef __attribute__((ext_vector_type(4)))  int          i32x4;
typedef __attribute__((ext_vector_type(8)))  int          i32x8;
typedef __attribute__((ext_vector_type(16))) __bf16       v16bf;
typedef __attribute__((ext_vector_type(8)))  float        v8f;

union Frag { u32x4 u[2]; v16bf v; };

__device__ __forceinline__ u16 f2bf(float x) {
  unsigned int u = __float_as_uint(x);
  u += 0x7FFFu + ((u >> 16) & 1u);   // round-to-nearest-even
  return (u16)(u >> 16);
}

__device__ __forceinline__ v8f wmma_bf16(const Frag& a, const Frag& b, v8f c) {
  return __builtin_amdgcn_wmma_f32_16x16x32_bf16(false, a.v, false, b.v,
                                                 (short)0, c, false, false);
}

// ---------------- elementwise prep kernels ----------------

__global__ void cast_bf16_kernel(const float* __restrict__ in,
                                 u16* __restrict__ out, int n) {
  int idx = blockIdx.x * blockDim.x + threadIdx.x;
  if (idx < n) out[idx] = f2bf(in[idx]);
}

// W is [K,N] fp32 row-major; WT becomes [N,K] bf16 row-major.
__global__ void transpose_cast_w_kernel(const float* __restrict__ W,
                                        u16* __restrict__ WT, int K, int N) {
  int idx = blockIdx.x * blockDim.x + threadIdx.x;
  if (idx >= N * K) return;
  int k = idx % K;
  int n = idx / K;
  WT[idx] = f2bf(W[(size_t)k * N + n]);
}

// RoPE: X fp32 [B, S, NH*HD] -> Out bf16 [B, NH, S, HD], interleaved pairs.
__global__ void rope_kernel(const float* __restrict__ X,
                            const float* __restrict__ cs,
                            const float* __restrict__ sn,
                            u16* __restrict__ Out, int NH) {
  int idx = blockIdx.x * blockDim.x + threadIdx.x;
  int total = B_ * NH * S_ * (HD_ / 2);
  if (idx >= total) return;
  int i = idx & 63;                 // pair index 0..63
  int s = (idx >> 6) & (S_ - 1);
  int h = (idx >> 17) % NH;
  int b = idx / (NH << 17);
  const float* xp = X + ((size_t)(b * S_ + s)) * (NH * HD_) + h * HD_ + 2 * i;
  float t0 = xp[0], t1 = xp[1];
  float c = cs[s * 64 + i], si = sn[s * 64 + i];
  size_t ob = ((size_t)(b * NH + h) * S_ + s) * HD_ + 2 * i;
  Out[ob]     = f2bf(t0 * c - t1 * si);
  Out[ob + 1] = f2bf(t0 * si + t1 * c);
}

// V fp32 [B, S, KVH*HD] -> Vt bf16 [B, KVH, HD, S]
__global__ void transpose_v_kernel(const float* __restrict__ Vf,
                                   u16* __restrict__ Vt) {
  int idx = blockIdx.x * blockDim.x + threadIdx.x;
  if (idx >= B_ * KVH_ * S_ * HD_) return;
  int hd = idx & (HD_ - 1);
  int s  = (idx >> 7) & (S_ - 1);
  int h  = (idx >> 18) & (KVH_ - 1);
  int b  = idx >> 21;
  float v = Vf[((size_t)(b * S_ + s)) * (KVH_ * HD_) + h * HD_ + hd];
  Vt[((size_t)((b * KVH_ + h) * HD_ + hd)) * S_ + s] = f2bf(v);
}

// ---------------- WMMA GEMM: C[M,N] = A[M,K](bf16) @ BT[N,K](bf16), fp32 out
// One wave computes a 32x64 tile; register double-buffered K pipeline so the
// compiler can issue partial s_wait_loadcnt instead of full stalls.
// __launch_bounds__(256, 1): allow the full tile + pipeline to live in VGPRs
// (no inner-loop scratch spills; latency is hidden by the register pipeline).

__global__ void __launch_bounds__(256, 1)
gemm_bf16_kernel(const u16* __restrict__ A,
                 const u16* __restrict__ BT,
                 float* __restrict__ C,
                 int M, int N, int K) {
  int wave = (blockIdx.x * blockDim.x + threadIdx.x) >> 5;
  int lane = threadIdx.x & 31;
  int n64  = N >> 6;
  int mt   = wave / n64;
  int nt   = wave % n64;
  if (mt >= (M >> 5)) return;

  int m    = lane & 15;
  int half = lane >> 4;

  const u16* arow0 = A + (size_t)(mt * 32 + m) * K + half * 8;
  const u16* arow1 = arow0 + (size_t)16 * K;
  const u16* brow[4];
#pragma unroll
  for (int nn = 0; nn < 4; ++nn)
    brow[nn] = BT + (size_t)(nt * 64 + nn * 16 + m) * K + half * 8;

  v8f acc[2][4];
#pragma unroll
  for (int i = 0; i < 2; ++i)
#pragma unroll
    for (int nn = 0; nn < 4; ++nn) acc[i][nn] = (v8f){};

  Frag aA[2], bA[4], aB[2], bB[4];

  auto load_step = [&](Frag* fa, Frag* fb, int k0) {
    fa[0].u[0] = *(const u32x4*)(arow0 + k0);
    fa[0].u[1] = *(const u32x4*)(arow0 + k0 + 16);
    fa[1].u[0] = *(const u32x4*)(arow1 + k0);
    fa[1].u[1] = *(const u32x4*)(arow1 + k0 + 16);
#pragma unroll
    for (int nn = 0; nn < 4; ++nn) {
      fb[nn].u[0] = *(const u32x4*)(brow[nn] + k0);
      fb[nn].u[1] = *(const u32x4*)(brow[nn] + k0 + 16);
    }
  };
  auto compute_step = [&](Frag* fa, Frag* fb) {
#pragma unroll
    for (int i = 0; i < 2; ++i)
#pragma unroll
      for (int nn = 0; nn < 4; ++nn)
        acc[i][nn] = wmma_bf16(fa[i], fb[nn], acc[i][nn]);
  };

  load_step(aA, bA, 0);
  int k0 = 0;
  for (; k0 < K - 64; k0 += 64) {
    load_step(aB, bB, k0 + 32);          // in flight while computing k0
    __builtin_prefetch(arow0 + k0 + 256, 0, 1);
    compute_step(aA, bA);
    load_step(aA, bA, k0 + 64);          // in flight while computing k0+32
    compute_step(aB, bB);
  }
  load_step(aB, bB, k0 + 32);            // k0 == K-64 here
  compute_step(aA, bA);
  compute_step(aB, bB);

#pragma unroll
  for (int i = 0; i < 2; ++i)
#pragma unroll
    for (int nn = 0; nn < 4; ++nn)
#pragma unroll
      for (int j = 0; j < 8; ++j) {
        int row = mt * 32 + i * 16 + half * 8 + j;
        int col = nt * 64 + nn * 16 + m;
        C[(size_t)row * N + col] = acc[i][nn][j];
      }
}

// ---------------- flash attention, one wave per 16-query tile --------------
// 4 waves / block. K tiles (32 keys x 128 hd, bf16) are staged into LDS by the
// Tensor Data Mover (TENSOR_LOAD_TO_LDS + s_wait_tensorcnt), then consumed as
// WMMA B fragments via ds_load_b128.

__global__ void __launch_bounds__(128, 1)
attn_kernel(const u16* __restrict__ Qb,
            const u16* __restrict__ Kb,
            const u16* __restrict__ Vt,
            u16* __restrict__ Ob) {
  __shared__ __align__(16) u16 kpanel[4][32 * HD_];   // 32 KB
  __shared__ __align__(16) u16 plds[4][16 * 32];      // 4 KB
  int wid  = threadIdx.x >> 5;
  int lane = threadIdx.x & 31;
  int gw   = blockIdx.x * 4 + wid;            // B_*H_*(S_/16) waves
  int qt   = gw % (S_ / 16);
  int bh   = gw / (S_ / 16);
  int h    = bh % H_;
  int b    = bh / H_;
  int kvh  = h >> 1;                          // n_rep = 2
  int q0   = qt * 16;
  int m    = lane & 15;
  int half = lane >> 4;

  // Q fragments: 4 chunks of K=32 over HD=128
  const u16* qrow = Qb + ((size_t)(b * H_ + h) * S_ + q0 + m) * HD_ + half * 16;
  Frag fq[4];
#pragma unroll
  for (int c = 0; c < 4; ++c) {
    fq[c].u[0] = *(const u32x4*)(qrow + c * 32);
    fq[c].u[1] = *(const u32x4*)(qrow + c * 32 + 8);
  }

  v8f o[8];
#pragma unroll
  for (int c = 0; c < 8; ++c) o[c] = (v8f){};
  float mrow[8], lrow[8];
#pragma unroll
  for (int j = 0; j < 8; ++j) { mrow[j] = -1e30f; lrow[j] = 0.0f; }

  const u16* kbase = Kb + (size_t)(b * KVH_ + kvh) * S_ * HD_;
  const u16* vbase = Vt + (size_t)(b * KVH_ + kvh) * HD_ * S_;
  const float sc = 0.08838834764831845f;      // 1/sqrt(128)

  unsigned lds_off = (unsigned)(uintptr_t)(&kpanel[wid][0]);

  int nkb = (q0 + 16 + 31) >> 5;              // 32-key blocks (causal bound)
  for (int kb = 0; kb < nkb; ++kb) {
    int k0 = kb * 32;

    // ---- TDM: stage K tile [32 keys x 128 hd] (bf16) into LDS ----
    {
      unsigned long long ga =
          (unsigned long long)(uintptr_t)(kbase + (size_t)k0 * HD_);
      u32x4 g0;
      g0[0] = 1u;                                        // count=1
      g0[1] = lds_off;                                   // lds_addr
      g0[2] = (unsigned)(ga & 0xFFFFFFFFu);              // global_addr lo
      g0[3] = (unsigned)((ga >> 32) & 0x1FFFFFFu) | 0x80000000u; // hi | type=2
      i32x8 g1;
      g1[0] = 0x00010000;          // workgroup_mask=0, data_size=1 (2 bytes)
      g1[1] = (int)((HD_ & 0xFFFF) << 16);               // tensor_dim0 lo16
      g1[2] = (int)((HD_ >> 16) | ((S_ & 0xFFFF) << 16));// dim0 hi | dim1 lo
      g1[3] = (int)((S_ >> 16) | (HD_ << 16));           // dim1 hi | tile_dim0
      g1[4] = 32;                                        // tile_dim1=32
      g1[5] = HD_;                                       // dim0_stride lo32
      g1[6] = 0;                                         // stride0 hi | s1 lo
      g1[7] = 0;
      i32x4 gz4 = {0, 0, 0, 0};
      i32x8 gz8 = {0, 0, 0, 0, 0, 0, 0, 0};
      __builtin_amdgcn_tensor_load_to_lds(g0, g1, gz4, gz4, gz8, 0);
      __builtin_amdgcn_s_wait_tensorcnt((short)0);
      asm volatile("" ::: "memory");   // TDM wrote LDS behind compiler's back
    }

    // ---- scores: Q @ K^T from the LDS panel ----
    v8f s0 = (v8f){}, s1 = (v8f){};
    const u16* kp0 = &kpanel[wid][(size_t)m * HD_ + half * 16];
    const u16* kp1 = kp0 + 16 * HD_;
#pragma unroll
    for (int c = 0; c < 4; ++c) {
      Frag fk0, fk1;
      fk0.u[0] = *(const u32x4*)(kp0 + c * 32);
      fk0.u[1] = *(const u32x4*)(kp0 + c * 32 + 8);
      fk1.u[0] = *(const u32x4*)(kp1 + c * 32);
      fk1.u[1] = *(const u32x4*)(kp1 + c * 32 + 8);
      s0 = wmma_bf16(fq[c], fk0, s0);
      s1 = wmma_bf16(fq[c], fk1, s1);
    }

    // ---- online softmax over this 16x32 block ----
#pragma unroll
    for (int j = 0; j < 8; ++j) {
      int row = q0 + half * 8 + j;
      int c0  = k0 + m;
      int c1  = c0 + 16;
      float v0 = (c0 > row) ? -1e30f : s0[j] * sc;
      float v1 = (c1 > row) ? -1e30f : s1[j] * sc;
      float mx = fmaxf(v0, v1);
#pragma unroll
      for (int off = 8; off >= 1; off >>= 1)
        mx = fmaxf(mx, __shfl_xor(mx, off, 32));
      float mnew  = fmaxf(mrow[j], mx);
      float alpha = __expf(mrow[j] - mnew);
      float p0 = __expf(v0 - mnew);
      float p1 = __expf(v1 - mnew);
      float rs = p0 + p1;
#pragma unroll
      for (int off = 8; off >= 1; off >>= 1)
        rs += __shfl_xor(rs, off, 32);
      lrow[j] = lrow[j] * alpha + rs;
      mrow[j] = mnew;
#pragma unroll
      for (int c = 0; c < 8; ++c) o[c][j] *= alpha;
      int rowm = half * 8 + j;
      plds[wid][rowm * 32 + m]      = f2bf(p0);
      plds[wid][rowm * 32 + 16 + m] = f2bf(p1);
    }

    // ---- P (16x32) as A fragment from LDS ----
    Frag fp;
    const u16* pr = &plds[wid][m * 32 + half * 16];
    fp.u[0] = *(const u32x4*)(pr);
    fp.u[1] = *(const u32x4*)(pr + 8);

    // ---- O += P @ V  (8 N-tiles of 16 over HD=128) ----
#pragma unroll
    for (int ct = 0; ct < 8; ++ct) {
      const u16* vrow = vbase + (size_t)(ct * 16 + m) * S_ + k0 + half * 16;
      Frag fv;
      fv.u[0] = *(const u32x4*)(vrow);
      fv.u[1] = *(const u32x4*)(vrow + 8);
      o[ct] = wmma_bf16(fp, fv, o[ct]);
    }
  }

  // normalize and store bf16 in [B*S, H*HD] layout for the output GEMM
#pragma unroll
  for (int j = 0; j < 8; ++j) {
    float inv = 1.0f / lrow[j];
    int row = q0 + half * 8 + j;
    size_t obase = ((size_t)b * S_ + row) * (H_ * HD_) + h * HD_;
#pragma unroll
    for (int ct = 0; ct < 8; ++ct)
      Ob[obase + ct * 16 + m] = f2bf(o[ct][j] * inv);
  }
}

// ---------------- launcher ----------------

extern "C" void kernel_launch(void* const* d_in, const int* in_sizes, int n_in,
                              void* d_out, int out_size, void* d_ws, size_t ws_size,
                              hipStream_t stream) {
  const float* x  = (const float*)d_in[0];
  const float* cs = (const float*)d_in[1];
  const float* sn = (const float*)d_in[2];
  const float* wq = (const float*)d_in[3];
  const float* wk = (const float*)d_in[4];
  const float* wv = (const float*)d_in[5];
  const float* wo = (const float*)d_in[6];
  float* out = (float*)d_out;
  char* ws = (char*)d_ws;

  const size_t M = (size_t)B_ * S_;                // 4096
  size_t off = 0;
  auto alloc = [&](size_t bytes) { size_t o = off; off += (bytes + 255) & ~(size_t)255; return o; };

  u16*   xb  = (u16*)(ws + alloc(M * D_ * 2));               // x bf16
  u16*   wqT = (u16*)(ws + alloc((size_t)D_ * (H_*HD_) * 2));
  u16*   wkT = (u16*)(ws + alloc((size_t)D_ * (KVH_*HD_) * 2));
  u16*   wvT = (u16*)(ws + alloc((size_t)D_ * (KVH_*HD_) * 2));
  u16*   woT = (u16*)(ws + alloc((size_t)(H_*HD_) * D_ * 2));
  float* Qf  = (float*)(ws + alloc(M * (H_*HD_) * 4));
  float* Kf  = (float*)(ws + alloc(M * (KVH_*HD_) * 4));
  float* Vf  = (float*)(ws + alloc(M * (KVH_*HD_) * 4));
  u16*   Qb  = (u16*)(ws + alloc(M * (H_*HD_) * 2));
  u16*   Kb  = (u16*)(ws + alloc(M * (KVH_*HD_) * 2));
  u16*   Vt  = (u16*)(ws + alloc(M * (KVH_*HD_) * 2));
  u16*   Ob  = (u16*)(ws + alloc(M * (H_*HD_) * 2));
  (void)alloc(4096);                                         // tail pad

  const int TB = 256;

  // 1) casts / transposes
  cast_bf16_kernel<<<(int)(M * D_ / TB), TB, 0, stream>>>(x, xb, (int)(M * D_));
  transpose_cast_w_kernel<<<(D_*(H_*HD_))/TB,   TB, 0, stream>>>(wq, wqT, D_, H_*HD_);
  transpose_cast_w_kernel<<<(D_*(KVH_*HD_))/TB, TB, 0, stream>>>(wk, wkT, D_, KVH_*HD_);
  transpose_cast_w_kernel<<<(D_*(KVH_*HD_))/TB, TB, 0, stream>>>(wv, wvT, D_, KVH_*HD_);
  transpose_cast_w_kernel<<<((H_*HD_)*D_)/TB,   TB, 0, stream>>>(wo, woT, H_*HD_, D_);

  // 2) QKV projections (fp32 out), 32x64 tile per wave
  {
    int waves = (int)(M / 32) * ((H_*HD_) / 64);
    gemm_bf16_kernel<<<waves / 8, TB, 0, stream>>>(xb, wqT, Qf, (int)M, H_*HD_, D_);
  }
  {
    int waves = (int)(M / 32) * ((KVH_*HD_) / 64);
    gemm_bf16_kernel<<<waves / 8, TB, 0, stream>>>(xb, wkT, Kf, (int)M, KVH_*HD_, D_);
    gemm_bf16_kernel<<<waves / 8, TB, 0, stream>>>(xb, wvT, Vf, (int)M, KVH_*HD_, D_);
  }

  // 3) RoPE + layout changes (bf16 out)
  rope_kernel<<<(B_*H_*S_*64)/TB,   TB, 0, stream>>>(Qf, cs, sn, Qb, H_);
  rope_kernel<<<(B_*KVH_*S_*64)/TB, TB, 0, stream>>>(Kf, cs, sn, Kb, KVH_);
  transpose_v_kernel<<<(B_*KVH_*S_*HD_)/TB, TB, 0, stream>>>(Vf, Vt);

  // 4) causal flash attention (TDM-staged K tiles), 4 waves/block
  attn_kernel<<<(B_*H_*(S_/16))/4, 128, 0, stream>>>(Qb, Kb, Vt, Ob);

  // 5) output projection straight into d_out (fp32)
  {
    int waves = (int)(M / 32) * (D_ / 64);
    gemm_bf16_kernel<<<waves / 8, TB, 0, stream>>>(Ob, woT, out, (int)M, D_, H_*HD_);
  }
}
